// Dynamic_21801253994880
// MI455X (gfx1250) — matrix operand order repaired and checked by reading, not compiled, not run on hardware
//
#include <hip/hip_runtime.h>

// ---------------- problem constants ----------------
#define BATCH 16
#define NNODE 512
#define DIM   32
#define EPG   8176                 // 16*(512-1)
#define NTOT  (BATCH*NNODE)        // 8192
#define TOTE  (BATCH*EPG)          // 130816
#define FSSZ  601
#define SLEFT (NNODE*DIM)          // 16384
#define NSROW (SLEFT + 2*EPG)      // 32736
#define LDA   640                  // fixed activation row stride (floats)

typedef __attribute__((ext_vector_type(8)))  float  v8f;
typedef __attribute__((ext_vector_type(8)))  __bf16 v8bf;
typedef __attribute__((ext_vector_type(16))) __bf16 v16bf;

// fp32 -> bf16 (round-to-nearest-even), stored as ushort
__device__ __forceinline__ unsigned short f2bf_u(float f) {
  unsigned u = __builtin_bit_cast(unsigned, f);
  return (unsigned short)((u + 0x7FFFu + ((u >> 16) & 1u)) >> 16);
}
// ordered-uint encoding for float atomicMax
__device__ __forceinline__ unsigned fenc(float f) {
  unsigned u = __builtin_bit_cast(unsigned, f);
  return (u & 0x80000000u) ? ~u : (u | 0x80000000u);
}
__device__ __forceinline__ float fdec(unsigned e) {
  unsigned u = (e & 0x80000000u) ? (e & 0x7FFFFFFFu) : ~e;
  return __builtin_bit_cast(float, u);
}

// ---------------- small utility kernels ----------------
__global__ void k_zero(float* p, int n) {
  int i = blockIdx.x * 256 + threadIdx.x;
  if (i < n) p[i] = 0.f;
}

__global__ void k_build_idx(const float* __restrict__ ns, int* __restrict__ snd,
                            int* __restrict__ rcv) {
  int t = blockIdx.x * 256 + threadIdx.x;           // < TOTE
  int g = t / EPG, e = t % EPG;
  const float* row = ns + (size_t)g * NSROW + SLEFT;
  snd[t] = (int)row[e]       + g * NNODE;
  rcv[t] = (int)row[EPG + e] + g * NNODE;
}

// sa = [nodes | one_hot(a)]  -> X cols 0..32 (rest already zeroed)
__global__ void k_build_sa(const float* __restrict__ ns, const int* __restrict__ a,
                           float* __restrict__ X) {
  int t = blockIdx.x * 256 + threadIdx.x;           // NTOT*33
  if (t >= NTOT * 33) return;
  int row = t / 33, c = t % 33;
  int g = row >> 9, k = row & 511;
  float v = (c < DIM) ? ns[(size_t)g * NSROW + k * DIM + c]
                      : ((a[g] == k) ? 1.f : 0.f);
  X[(size_t)row * LDA + c] = v;
}

// W [din,dout] fp32 -> Wt [doutPadB, dinPad] bf16 (transposed + zero padded)
__global__ void k_prep_w(const float* __restrict__ W, unsigned short* __restrict__ Wt,
                         int din, int dout, int dinPad, int doutPadB) {
  int t = blockIdx.x * 256 + threadIdx.x;
  if (t >= doutPadB * dinPad) return;
  int j = t / dinPad, k = t % dinPad;
  float v = (j < dout && k < din) ? W[(size_t)k * dout + j] : 0.f;
  Wt[(size_t)j * dinPad + k] = f2bf_u(v);
}

// ---------------- WMMA GEMM: H[8192, doutPadB] = X[8192, K] @ Wt^T ----------------
// block: 256 thr (8 waves) -> 64x128 tile; wave -> 32x32 (2x2 WMMA accum)
// B tile staged with CDNA5 async global->LDS DMA (ASYNCcnt), A tile converted fp32->bf16.
__global__ __launch_bounds__(256) void k_gemm(const float* __restrict__ X,
                                              const unsigned short* __restrict__ Wt,
                                              float* __restrict__ H, int K) {
  __shared__ __attribute__((aligned(32))) unsigned short At[64][32];   // A tile, row-major
  __shared__ __attribute__((aligned(32))) unsigned short Bt[128][32];  // B^T tile: [n][k]
  const int m0 = blockIdx.x * 64;
  const int n0 = blockIdx.y * 128;
  const int t    = threadIdx.x;
  const int lane = t & 31, wave = t >> 5;
  const int wm = (wave >> 2) * 32;        // 0,32
  const int wn = (wave & 3) * 32;         // 0,32,64,96
  const int r  = lane & 15;
  const int hh = lane >> 4;               // lane half

  // per-thread staging coordinates (fixed across K loop)
  const int arow = t >> 2, ac0 = (t & 3) * 8;      // A: 8 floats
  const int bc = t >> 1, bk0 = (t & 1) * 16;       // B: 32 bytes (2 x b128)
  const unsigned bLds = (unsigned)(size_t)&Bt[bc][bk0];

  v8f acc[2][2] = {};

  for (int kk = 0; kk < K; kk += 32) {
    __syncthreads();
    {   // stage B^T via async global->LDS DMA: 128 cols x 32 k bf16 (8KB/tile)
      const unsigned short* gp = Wt + (size_t)(n0 + bc) * K + kk + bk0;
      asm volatile(
          "global_load_async_to_lds_b128 %0, %1, off\n\t"
          "global_load_async_to_lds_b128 %0, %1, off offset:16"
          :: "v"(bLds), "v"(gp) : "memory");
    }
    {   // stage A: 64x32 fp32 -> bf16 (needs VALU convert, so VGPR path)
      const float* src = X + (size_t)(m0 + arow) * LDA + kk + ac0;
#pragma unroll
      for (int i = 0; i < 8; ++i) At[arow][ac0 + i] = f2bf_u(src[i]);
      // prefetch next K panel of A into cache while WMMAs run
      __builtin_prefetch(src + 32, 0, 1);
    }
    asm volatile("s_wait_asynccnt 0" ::: "memory");
    __syncthreads();

    v16bf af[2], bfr[2];
#pragma unroll
    for (int i = 0; i < 2; ++i) {   // A frag: lane holds row (wm+i*16+r), 16 contiguous K
      af[i] = *(const v16bf*)&At[wm + i * 16 + r][hh * 16];
    }
#pragma unroll
    for (int j = 0; j < 2; ++j)     // B frag: lane holds col (wn+j*16+r), 16 contiguous K
      bfr[j] = *(const v16bf*)&Bt[wn + j * 16 + r][hh * 16];

#pragma unroll
    for (int i = 0; i < 2; ++i)
#pragma unroll
      for (int j = 0; j < 2; ++j)
        acc[i][j] = __builtin_amdgcn_wmma_f32_16x16x32_bf16(
            false, af[i], false, bfr[j], (short)0, acc[i][j], false, false);
  }

#pragma unroll
  for (int i = 0; i < 2; ++i)
#pragma unroll
    for (int j = 0; j < 2; ++j)
#pragma unroll
      for (int v = 0; v < 8; ++v) {
        int row = m0 + wm + i * 16 + hh * 8 + v;   // C layout: VGPR v -> M=v (+8 for hi half)
        int col = n0 + wn + j * 16 + r;
        H[(size_t)row * LDA + col] = acc[i][j][v];
      }
}

// ---------------- attention / softmax / aggregation ----------------
__global__ void k_scores(const float* __restrict__ H, const float* __restrict__ as,
                         const float* __restrict__ ad, float* __restrict__ es,
                         float* __restrict__ ed, int dout) {
  int wid  = (blockIdx.x * 256 + threadIdx.x) >> 5;
  int lane = threadIdx.x & 31;
  if (wid >= NTOT) return;
  const float* h = H + (size_t)wid * LDA;
  float s = 0.f, d = 0.f;
  for (int j = lane; j < dout; j += 32) { float v = h[j]; s += v * as[j]; d += v * ad[j]; }
  for (int o = 16; o; o >>= 1) { s += __shfl_down(s, o, 32); d += __shfl_down(d, o, 32); }
  if (!lane) { es[wid] = s; ed[wid] = d; }
}

__global__ void k_init_nodes(unsigned* __restrict__ mb, float* __restrict__ den) {
  int i = blockIdx.x * 256 + threadIdx.x;
  if (i < NTOT) { mb[i] = 0x007FFFFFu /* enc(-inf) */; den[i] = 0.f; }
}

__global__ void k_init_agg(float* __restrict__ agg, const float* __restrict__ b, int dout) {
  int t = blockIdx.x * 256 + threadIdx.x;
  if (t >= NTOT * dout) return;
  int n = t / dout, j = t % dout;
  agg[(size_t)n * LDA + j] = b[j];
}

__global__ void k_edge_max(const float* __restrict__ es, const float* __restrict__ ed,
                           const int* __restrict__ snd, const int* __restrict__ rcv,
                           float* __restrict__ elog, unsigned* __restrict__ mb) {
  int t = blockIdx.x * 256 + threadIdx.x;
  if (t >= TOTE) return;
  float e = es[snd[t]] + ed[rcv[t]];
  e = e > 0.f ? e : 0.2f * e;                       // LeakyReLU(0.2)
  elog[t] = e;
  atomicMax(mb + rcv[t], fenc(e));
}

__global__ void k_fix_m(const unsigned* __restrict__ mb, float* __restrict__ mf) {
  int i = blockIdx.x * 256 + threadIdx.x;
  if (i >= NTOT) return;
  float m = fdec(mb[i]);
  mf[i] = (m >= -3.0e38f && m <= 3.0e38f) ? m : 0.f;
}

__global__ void k_edge_exp(const float* __restrict__ elog, const float* __restrict__ mf,
                           const int* __restrict__ rcv, float* __restrict__ ee,
                           float* __restrict__ den) {
  int t = blockIdx.x * 256 + threadIdx.x;
  if (t >= TOTE) return;
  float v = __expf(elog[t] - mf[rcv[t]]);
  ee[t] = v;
  atomicAdd(den + rcv[t], v);
}

// wave per edge: agg[rcv] += alpha * h[snd]
__global__ void k_aggregate(const float* __restrict__ ee, const float* __restrict__ den,
                            const float* __restrict__ H, const int* __restrict__ snd,
                            const int* __restrict__ rcv, float* __restrict__ agg, int dout) {
  int wid  = (blockIdx.x * 256 + threadIdx.x) >> 5;
  int lane = threadIdx.x & 31;
  if (wid >= TOTE) return;
  int s = snd[wid], rv = rcv[wid];
  float alpha = ee[wid] / (den[rv] + 1e-16f);
  const float* h = H + (size_t)s * LDA;
  float* o = agg + (size_t)rv * LDA;
  for (int j = lane; j < dout; j += 32) atomicAdd(o + j, alpha * h[j]);
}

// LayerNorm over node axis (512 nodes, 16 per lane on wave32) + ReLU -> X
__global__ void k_ln_relu(const float* __restrict__ agg, const float* __restrict__ sc,
                          const float* __restrict__ of, float* __restrict__ X, int dout) {
  int wid  = (blockIdx.x * 256 + threadIdx.x) >> 5;
  int lane = threadIdx.x & 31;
  if (wid >= BATCH * dout) return;
  int g = wid / dout, j = wid % dout;
  const float* base = agg + ((size_t)g * NNODE) * LDA + j;
  float x[16], s = 0.f;
#pragma unroll
  for (int i = 0; i < 16; ++i) { x[i] = base[(size_t)(lane + 32 * i) * LDA]; s += x[i]; }
  for (int o = 16; o; o >>= 1) s += __shfl_down(s, o, 32);
  float mu = __shfl(s, 0, 32) * (1.f / NNODE);
  float vs = 0.f;
#pragma unroll
  for (int i = 0; i < 16; ++i) { float d = x[i] - mu; vs += d * d; }
  for (int o = 16; o; o >>= 1) vs += __shfl_down(vs, o, 32);
  float var = __shfl(vs, 0, 32) * (1.f / NNODE);
  float rs = rsqrtf(var + 1e-5f);
  float scale = sc[j], off = of[j];
  float* out = X + ((size_t)g * NNODE) * LDA + j;
#pragma unroll
  for (int i = 0; i < 16; ++i) {
    float y = (x[i] - mu) * rs * scale + off;
    out[(size_t)(lane + 32 * i) * LDA] = y > 0.f ? y : 0.f;
  }
}

// reward: r[g,j] = sum over nodes of agg
__global__ void k_reduce_r(const float* __restrict__ agg, float* __restrict__ out) {
  int wid  = (blockIdx.x * 256 + threadIdx.x) >> 5;
  int lane = threadIdx.x & 31;
  if (wid >= BATCH * FSSZ) return;
  int g = wid / FSSZ, j = wid % FSSZ;
  const float* base = agg + ((size_t)g * NNODE) * LDA + j;
  float s = 0.f;
#pragma unroll
  for (int i = 0; i < 16; ++i) s += base[(size_t)(lane + 32 * i) * LDA];
  for (int o = 16; o; o >>= 1) s += __shfl_down(s, o, 32);
  if (!lane) out[(size_t)g * FSSZ + j] = s;
}

__global__ void k_write_f(const float* __restrict__ agg, float* __restrict__ out) {
  int t = blockIdx.x * 256 + threadIdx.x;           // NTOT*DIM
  if (t >= NTOT * DIM) return;
  int n = t / DIM, d = t % DIM;
  int g = n >> 9, k = n & 511;
  out[(size_t)g * NSROW + k * DIM + d] = agg[(size_t)n * LDA + d];
}

__global__ void k_copy_tail(const float* __restrict__ ns, float* __restrict__ out) {
  int t = blockIdx.x * 256 + threadIdx.x;           // BATCH*2*EPG
  if (t >= BATCH * 2 * EPG) return;
  int g = t / (2 * EPG), e = t % (2 * EPG);
  out[(size_t)g * NSROW + SLEFT + e] = ns[(size_t)g * NSROW + SLEFT + e];
}

// ---------------- host driver ----------------
extern "C" void kernel_launch(void* const* d_in, const int* in_sizes, int n_in,
                              void* d_out, int out_size, void* d_ws, size_t ws_size,
                              hipStream_t stream) {
  (void)in_sizes; (void)n_in; (void)out_size; (void)ws_size;
  const float* ns = (const float*)d_in[0];
  const int*   a  = (const int*)d_in[1];

  char* w = (char*)d_ws;
  auto alloc = [&](size_t bytes) { char* p = w; w += (bytes + 255) & ~(size_t)255; return p; };
  int*            SND  = (int*)alloc((size_t)TOTE * 4);
  int*            RCV  = (int*)alloc((size_t)TOTE * 4);
  float*          X    = (float*)alloc((size_t)NTOT * LDA * 4);
  float*          H    = (float*)alloc((size_t)NTOT * LDA * 4);
  float*          AGG  = (float*)alloc((size_t)NTOT * LDA * 4);
  unsigned short* WT   = (unsigned short*)alloc((size_t)640 * 640 * 2);
  float*          ES   = (float*)alloc((size_t)NTOT * 4);
  float*          ED   = (float*)alloc((size_t)NTOT * 4);
  unsigned*       MB   = (unsigned*)alloc((size_t)NTOT * 4);
  float*          MF   = (float*)alloc((size_t)NTOT * 4);
  float*          DEN  = (float*)alloc((size_t)NTOT * 4);
  float*          ELOG = (float*)alloc((size_t)TOTE * 4);
  float*          EE   = (float*)alloc((size_t)TOTE * 4);

  k_build_idx<<<TOTE / 256, 256, 0, stream>>>(ns, SND, RCV);

  const int dinT[2][4]  = {{33, 601, 601, 601}, {33, 128, 64, 64}};
  const int doutT[2][4] = {{601, 601, 601, 601}, {128, 64, 64, 32}};
  const int baseIdx[2]  = {24, 2};   // head 0 = reward (rW*), head 1 = next-state (W*)
  float* dr = (float*)d_out;         // r: [16,601]
  float* df = dr + BATCH * FSSZ;     // ns_new: [16,32736]

  for (int head = 0; head < 2; ++head) {
    k_zero<<<(NTOT * LDA) / 256, 256, 0, stream>>>(X, NTOT * LDA);
    k_build_sa<<<(NTOT * 33 + 255) / 256, 256, 0, stream>>>(ns, a, X);
    int pi = baseIdx[head];
    for (int L = 0; L < 4; ++L) {
      int din = dinT[head][L], dout = doutT[head][L];
      int dinPad  = (din + 31) & ~31;
      int doutPB  = (dout + 127) & ~127;
      const float* W  = (const float*)d_in[pi++];
      const float* as = (const float*)d_in[pi++];
      const float* ad = (const float*)d_in[pi++];
      const float* b  = (const float*)d_in[pi++];
      const float* sc = nullptr; const float* of = nullptr;
      if (L < 3) { sc = (const float*)d_in[pi++]; of = (const float*)d_in[pi++]; }

      k_prep_w<<<((size_t)doutPB * dinPad + 255) / 256, 256, 0, stream>>>(
          W, WT, din, dout, dinPad, doutPB);
      dim3 gg(NTOT / 64, doutPB / 128);
      k_gemm<<<gg, 256, 0, stream>>>(X, WT, H, dinPad);

      k_scores<<<NTOT / 8, 256, 0, stream>>>(H, as, ad, ES, ED, dout);
      k_init_nodes<<<NTOT / 256, 256, 0, stream>>>(MB, DEN);
      k_init_agg<<<((size_t)NTOT * dout + 255) / 256, 256, 0, stream>>>(AGG, b, dout);
      k_edge_max<<<TOTE / 256, 256, 0, stream>>>(ES, ED, SND, RCV, ELOG, MB);
      k_fix_m<<<NTOT / 256, 256, 0, stream>>>(MB, MF);
      k_edge_exp<<<TOTE / 256, 256, 0, stream>>>(ELOG, MF, RCV, EE, DEN);
      k_aggregate<<<TOTE / 8, 256, 0, stream>>>(EE, DEN, H, SND, RCV, AGG, dout);

      if (L < 3)
        k_ln_relu<<<(BATCH * dout + 7) / 8, 256, 0, stream>>>(AGG, sc, of, X, dout);
    }
    if (head == 0) {
      k_reduce_r<<<(BATCH * FSSZ + 7) / 8, 256, 0, stream>>>(AGG, dr);
    } else {
      k_write_f<<<(NTOT * DIM) / 256, 256, 0, stream>>>(AGG, df);
      k_copy_tail<<<(BATCH * 2 * EPG) / 256, 256, 0, stream>>>(ns, df);
    }
  }
}
